// MinkEncoderDecoder_24962349924883
// MI455X (gfx1250) — matrix operand order repaired and checked by reading, not compiled, not run on hardware
//
#include <hip/hip_runtime.h>

// =====================================================================
// MinkEncoderDecoder on MI455X (gfx1250, wave32, WMMA + async LDS gather)
//
// Input flattening assumption (depth-first, dict insertion order):
//   0: feats (N0 x 4 f32)
//   stem: 1 W1(27,4,32) 2 bn1.g 3 bn1.b 4 W2(27,32,32) 5 bn2.g 6 bn2.b
//   s1 @7, s2 @25, s3 @43, s4 @61, each:
//     +0 Wdown(8,ci,ci) +1 bndown.g +2 bndown.b
//     res1 @+3: W1 bn1g bn1b W2 bn2g bn2b [Wd bndg bndb]   (Wd only if ci!=co)
//     res2 @(+12 or +9): W1 bn1g bn1b W2 bn2g bn2b
//   nbrs: 76..80 n3_0..n3_4 (Nl x 27 i32), 81..84 down_0..3 (N(l+1) x 8 i32)
//   85: bid4 (N4 i32)
// Output: (2,256) f32 segment means.
// =====================================================================

// Flip to 0 if the assembler rejects the async-to-LDS mnemonics.
#define USE_ASYNC_GATHER 1

typedef __attribute__((ext_vector_type(16))) __bf16        v16bf;
typedef __attribute__((ext_vector_type(8)))  float         v8f;
typedef __attribute__((ext_vector_type(4)))  unsigned int  u32x4;

union ABfrag { v16bf v; u32x4 q[2]; };

__device__ __forceinline__ unsigned short f2bf(float f) {
  unsigned int u = __builtin_bit_cast(unsigned int, f);
  u += 0x7fffu + ((u >> 16) & 1u);            // round-to-nearest-even
  return (unsigned short)(u >> 16);
}
__device__ __forceinline__ float bf2f(unsigned short h) {
  unsigned int u = ((unsigned int)h) << 16;
  return __builtin_bit_cast(float, u);
}

// ---------------------------------------------------------------------
// Pack fp32 weights [Kt][CiReal][Co] into WMMA B-fragment layout, bf16.
// Tile (k, cb, cot) = 32x16 (Kdim x N); per lane: col = lane%16,
// kbase = cb*32 + (lane<16?0:8); dword j holds bf16 pair (kbase+2j+(j<4?0:8), +1).
// 256 dwords per tile.
// ---------------------------------------------------------------------
__global__ void pack_w(const float* __restrict__ W, unsigned int* __restrict__ out,
                       int Kt, int CiReal, int CiPad, int Co) {
  int cbN = CiPad >> 5, coN = Co >> 4;
  int total = Kt * cbN * coN * 256;
  int g = blockIdx.x * blockDim.x + threadIdx.x;
  if (g >= total) return;
  int t = g >> 8;
  int l = (g >> 3) & 31;
  int j = g & 7;
  int k   = t / (cbN * coN);
  int rem = t - k * (cbN * coN);
  int cb  = rem / coN;
  int cot = rem - cb * coN;
  int col = cot * 16 + (l & 15);
  int ci0 = cb * 32 + ((l < 16) ? 0 : 8) + 2 * j + ((j < 4) ? 0 : 8);
  float f0 = (ci0     < CiReal) ? W[((size_t)k * CiReal + ci0    ) * Co + col] : 0.0f;
  float f1 = (ci0 + 1 < CiReal) ? W[((size_t)k * CiReal + ci0 + 1) * Co + col] : 0.0f;
  out[g] = (unsigned int)f2bf(f0) | ((unsigned int)f2bf(f1) << 16);
}

// ---------------------------------------------------------------------
// Gather-GEMM sparse conv: fout[i, :] = sum_k fin[nbr[i,k], :] @ W[k]
// Block = 256 threads = 8 waves; block tile = 32 rows x (8 waves * 16 cols).
// Per (k, ci-chunk): 32x32 bf16 A-tile staged in double-buffered LDS via
// GLOBAL_LOAD_ASYNC_TO_LDS_B64 (ASYNCcnt) -- or VGPR path if disabled.
// Each wave: one B fragment feeds TWO v_wmma_f32_16x16x32_bf16 (rows 0-15
// and 16-31). Out-of-range rows gather from a zeroed row so EXEC stays
// all-ones for every WMMA. CBN = CiPad/32 is a template constant.
// ---------------------------------------------------------------------
template <int CBN>
__global__ __launch_bounds__(256) void conv_wmma(
    const unsigned short* __restrict__ fin, float* __restrict__ fout,
    const int* __restrict__ nbr, int Nout, int Kt, int CiPad, int Co,
    const u32x4* __restrict__ packedW, const unsigned short* __restrict__ zrow) {
  __shared__ __align__(16) unsigned char smA[2][32 * 80];  // 32 rows, 64B + 16B pad

  const int tid  = threadIdx.x;
  const int lane = tid & 31;
  const int wave = tid >> 5;
  const int rowBase = blockIdx.x * 32;
  const int coTiles = Co >> 4;
  const int ct = blockIdx.y * 8 + wave;
  const bool waveActive = (ct < coTiles);

  // gather mapping: thread t -> (row gr 0..31, 8-byte chunk gd8 0..7)
  const int gr  = tid >> 3;
  const int gd8 = tid & 7;
  const int grow = rowBase + gr;
  const bool inRange = (grow < Nout);

  const int arow = lane & 15;
  const int sel  = (lane < 16) ? 0 : 16;     // byte select within row
  const int T = Kt * CBN;

  v8f acc0 = {0.f, 0.f, 0.f, 0.f, 0.f, 0.f, 0.f, 0.f};
  v8f acc1 = {0.f, 0.f, 0.f, 0.f, 0.f, 0.f, 0.f, 0.f};

  auto issue_gather = [&](int it, unsigned char* buf) {
    int k  = it / CBN;                        // compile-time divisor
    int cb = it - k * CBN;
    int idx = inRange ? (nbr ? nbr[(size_t)grow * Kt + k] : grow) : -1;
    const unsigned short* src =
        (idx >= 0) ? (fin + (size_t)idx * CiPad + cb * 32) : zrow;
    unsigned long long ga = (unsigned long long)src + (unsigned long long)(gd8 * 8);
    unsigned char* dst = buf + gr * 80 + gd8 * 8;
#if USE_ASYNC_GATHER
    unsigned int ldsOff = (unsigned int)(size_t)dst;   // LDS addr = low 32 bits
    asm volatile("global_load_async_to_lds_b64 %0, %1, off"
                 :: "v"(ldsOff), "v"(ga)
                 : "memory");
#else
    *(unsigned long long*)dst =
        *(const unsigned long long*)(size_t)ga;
#endif
  };

  issue_gather(0, smA[0]);

#pragma unroll 2
  for (int it = 0; it < T; ++it) {
#if USE_ASYNC_GATHER
    asm volatile("s_wait_asynccnt 0x0" ::: "memory");  // my async writes landed
#endif
    __syncthreads();                                   // everyone's writes visible
    if (it + 1 < T) issue_gather(it + 1, smA[(it + 1) & 1]);

    const unsigned char* buf = smA[it & 1];
    ABfrag a0, a1;
    a0.q[0] = *(const u32x4*)(buf + arow * 80 + sel);
    a0.q[1] = *(const u32x4*)(buf + arow * 80 + 32 + sel);
    a1.q[0] = *(const u32x4*)(buf + (16 + arow) * 80 + sel);
    a1.q[1] = *(const u32x4*)(buf + (16 + arow) * 80 + 32 + sel);

    if (waveActive) {
      ABfrag b;
      const u32x4* wp = packedW + ((size_t)(it * coTiles + ct)) * 64
                        + (size_t)lane * 2;
      b.q[0] = wp[0];
      b.q[1] = wp[1];
      acc0 = __builtin_amdgcn_wmma_f32_16x16x32_bf16(
          false, a0.v, false, b.v, (short)0, acc0, false, false);
      acc1 = __builtin_amdgcn_wmma_f32_16x16x32_bf16(
          false, a1.v, false, b.v, (short)0, acc1, false, false);
    }
  }

  if (waveActive) {
    const int col = ct * 16 + arow;
    const int r0  = rowBase + ((lane < 16) ? 0 : 8);
#pragma unroll
    for (int v = 0; v < 8; ++v) {
      int r = r0 + v;
      if (r < Nout) fout[(size_t)r * Co + col] = acc0[v];
      int r2 = r + 16;
      if (r2 < Nout) fout[(size_t)r2 * Co + col] = acc1[v];
    }
  }
}

// ---------------------------------------------------------------------
// BN statistics: stats[c] += sum, stats[C+c] += sumsq  (fp32 atomics)
// ---------------------------------------------------------------------
__global__ __launch_bounds__(256) void bn_stats(const float* __restrict__ X,
                                                float* __restrict__ stats,
                                                int N, int C, int rowsPerBlock) {
  __shared__ float ssum[256], ssq[256];
  int t = threadIdx.x;
  ssum[t] = 0.f; ssq[t] = 0.f;
  __syncthreads();
  int rpi = 256 / C;                         // C in {32,64,128,256}
  int c  = t & (C - 1);
  int r0 = blockIdx.x * rowsPerBlock + t / C;
  int rEnd = (blockIdx.x + 1) * rowsPerBlock;
  if (rEnd > N) rEnd = N;
  float s = 0.f, q = 0.f;
  for (int r = r0; r < rEnd; r += rpi) {
    float v = X[(size_t)r * C + c];
    s += v; q += v * v;
  }
  atomicAdd(&ssum[c], s);
  atomicAdd(&ssq[c], q);
  __syncthreads();
  if (t < C) {
    atomicAdd(&stats[c],     ssum[c]);
    atomicAdd(&stats[C + c], ssq[c]);
  }
}

// ---------------------------------------------------------------------
// Fused BN apply + activation (+ optional residual) -> bf16 features.
// mode 0: relu                1: leaky-relu(0.01)
// mode 2: relu(bn(X)+bf16 shortcut)   3: relu(bn(X)+bn(Xs))
// ---------------------------------------------------------------------
__global__ void bn_apply(const float* __restrict__ X, const float* __restrict__ st,
                         const float* __restrict__ g, const float* __restrict__ b,
                         unsigned short* __restrict__ out, int N, int C, int mode,
                         const unsigned short* __restrict__ sBf,
                         const float* __restrict__ Xs, const float* __restrict__ stS,
                         const float* __restrict__ gs, const float* __restrict__ bs) {
  int i = blockIdx.x * blockDim.x + threadIdx.x;
  int total = N * C;
  if (i >= total) return;
  int c = i & (C - 1);                       // C is a power of two
  float invN = 1.0f / (float)N;
  float m   = st[c] * invN;
  float var = st[C + c] * invN - m * m;
  float sc  = g[c] * rsqrtf(var + 1e-5f);
  float v   = (X[i] - m) * sc + b[c];
  if (mode == 2) {
    v += bf2f(sBf[i]);
  } else if (mode == 3) {
    float ms   = stS[c] * invN;
    float vs   = stS[C + c] * invN - ms * ms;
    float scs  = gs[c] * rsqrtf(vs + 1e-5f);
    v += (Xs[i] - ms) * scs + bs[c];
  }
  if (mode == 1) v = (v > 0.f) ? v : 0.01f * v;
  else           v = fmaxf(v, 0.f);
  out[i] = f2bf(v);
}

__global__ void zero_f32(float* __restrict__ p, int n) {
  int i = blockIdx.x * blockDim.x + threadIdx.x;
  if (i < n) p[i] = 0.f;
}

// feats (N0 x 4 f32) -> bf16 padded to 32 channels
__global__ void cvt_feats(const float* __restrict__ feats,
                          unsigned short* __restrict__ out, int N0) {
  int i = blockIdx.x * blockDim.x + threadIdx.x;
  if (i >= N0 * 32) return;
  int r = i >> 5, c = i & 31;
  out[i] = f2bf(c < 4 ? feats[r * 4 + c] : 0.f);
}

__global__ void seg_cnt(const int* __restrict__ bid, float* __restrict__ cnt, int N) {
  int i = blockIdx.x * blockDim.x + threadIdx.x;
  if (i < N) atomicAdd(&cnt[bid[i]], 1.f);
}
__global__ void seg_acc(const unsigned short* __restrict__ F,
                        const int* __restrict__ bid, float* __restrict__ seg,
                        int N) {
  int i = blockIdx.x * blockDim.x + threadIdx.x;
  if (i >= N * 256) return;
  int r = i >> 8, c = i & 255;
  atomicAdd(&seg[(size_t)bid[r] * 256 + c], bf2f(F[i]));
}
__global__ void seg_fin(const float* __restrict__ seg, const float* __restrict__ cnt,
                        float* __restrict__ out, int C) {
  int i = blockIdx.x * blockDim.x + threadIdx.x;
  if (i < 2 * C) out[i] = seg[i] / cnt[i / C];
}

// =====================================================================
extern "C" void kernel_launch(void* const* d_in, const int* in_sizes, int n_in,
                              void* d_out, int out_size, void* d_ws, size_t ws_size,
                              hipStream_t stream) {
  (void)n_in; (void)out_size; (void)ws_size;
  auto P  = [&](int i) { return (const float*)d_in[i]; };
  auto Iv = [&](int i) { return (const int*)d_in[i]; };

  const int Cs[5] = {32, 64, 128, 256, 256};
  int N[5];
  for (int L = 0; L < 5; ++L) N[L] = in_sizes[76 + L] / 27;

  size_t fmax = 0;
  for (int L = 0; L < 5; ++L) {
    size_t s = (size_t)N[L] * Cs[L];
    if (s > fmax) fmax = s;
  }

  // workspace carve-out
  size_t cur_off = 0;
  auto wsAlloc = [&](size_t bytes) {
    cur_off = (cur_off + 255) & ~(size_t)255;
    size_t o = cur_off; cur_off += bytes; return o;
  };
  char* ws = (char*)d_ws;
  unsigned int*   packW  = (unsigned int*)  (ws + wsAlloc(27ull * 8 * 16 * 1024));
  unsigned short* F0     = (unsigned short*)(ws + wsAlloc(fmax * 2));
  unsigned short* F1     = (unsigned short*)(ws + wsAlloc(fmax * 2));
  float*          T1     = (float*)         (ws + wsAlloc(fmax * 4));
  float*          T2     = (float*)         (ws + wsAlloc(fmax * 4));
  float*          statsA = (float*)         (ws + wsAlloc(512 * 4));
  float*          statsB = (float*)         (ws + wsAlloc(512 * 4));
  float*          seg    = (float*)         (ws + wsAlloc(512 * 4));
  float*          cnt    = (float*)         (ws + wsAlloc(16 * 4));
  unsigned short* zrow   = (unsigned short*)(ws + wsAlloc(256));  // 64B zeroed row (+pad)

  zero_f32<<<dim3(1), dim3(64), 0, stream>>>((float*)zrow, 64);

  auto conv = [&](const unsigned short* fin, float* fout, const int* nbr,
                  int Nout, int Kt, int CiPad, int Co, const float* W, int CiReal) {
    int cbN = CiPad >> 5, coN = Co >> 4;
    int total = Kt * cbN * coN * 256;
    pack_w<<<dim3((total + 255) / 256), dim3(256), 0, stream>>>(W, packW, Kt, CiReal, CiPad, Co);
    dim3 g((unsigned)((Nout + 31) / 32), (unsigned)((coN + 7) / 8), 1);
    const u32x4* pw = (const u32x4*)packW;
    switch (cbN) {
      case 1: conv_wmma<1><<<g, dim3(256), 0, stream>>>(fin, fout, nbr, Nout, Kt, CiPad, Co, pw, zrow); break;
      case 2: conv_wmma<2><<<g, dim3(256), 0, stream>>>(fin, fout, nbr, Nout, Kt, CiPad, Co, pw, zrow); break;
      case 4: conv_wmma<4><<<g, dim3(256), 0, stream>>>(fin, fout, nbr, Nout, Kt, CiPad, Co, pw, zrow); break;
      default: conv_wmma<8><<<g, dim3(256), 0, stream>>>(fin, fout, nbr, Nout, Kt, CiPad, Co, pw, zrow); break;
    }
  };
  auto bnstats = [&](const float* X, int Nl, int C, float* st) {
    zero_f32<<<dim3(2), dim3(256), 0, stream>>>(st, 2 * C);
    const int rpb = 1024;
    bn_stats<<<dim3((Nl + rpb - 1) / rpb), dim3(256), 0, stream>>>(X, st, Nl, C, rpb);
  };
  auto apply = [&](const float* X, const float* st, const float* g, const float* b,
                   unsigned short* out, int Nl, int C, int mode,
                   const unsigned short* sBf, const float* Xs, const float* stS,
                   const float* gs, const float* bs) {
    int total = Nl * C;
    bn_apply<<<dim3((total + 255) / 256), dim3(256), 0, stream>>>(
        X, st, g, b, out, Nl, C, mode, sBf, Xs, stS, gs, bs);
  };

  unsigned short* fcur = F0;
  unsigned short* falt = F1;
  auto swapF = [&]() { unsigned short* t = fcur; fcur = falt; falt = t; };

  // ---- stem ----
  int N0 = N[0];
  cvt_feats<<<dim3((N0 * 32 + 255) / 256), dim3(256), 0, stream>>>(P(0), fcur, N0);
  conv(fcur, T1, Iv(76), N0, 27, 32, 32, P(1), 4);
  bnstats(T1, N0, 32, statsA);
  apply(T1, statsA, P(2), P(3), falt, N0, 32, 0, nullptr, nullptr, nullptr, nullptr, nullptr);
  swapF();
  conv(fcur, T1, Iv(76), N0, 27, 32, 32, P(4), 32);
  bnstats(T1, N0, 32, statsA);
  apply(T1, statsA, P(5), P(6), falt, N0, 32, 0, nullptr, nullptr, nullptr, nullptr, nullptr);
  swapF();

  // ---- stages ----
  const int base[4] = {7, 25, 43, 61};
  for (int s = 1; s <= 4; ++s) {
    int bs = base[s - 1];
    int ci = Cs[s - 1], co = Cs[s];
    int Nl = N[s];
    const int* nbrD = Iv(81 + (s - 1));
    const int* nbr3 = Iv(76 + s);

    // downsample conv (8 taps, ci->ci), leaky relu
    conv(fcur, T1, nbrD, Nl, 8, ci, ci, P(bs + 0), ci);
    bnstats(T1, Nl, ci, statsA);
    apply(T1, statsA, P(bs + 1), P(bs + 2), falt, Nl, ci, 1,
          nullptr, nullptr, nullptr, nullptr, nullptr);
    swapF();

    bool hasWd = (s < 4);
    int rb0 = bs + 3;
    int rb1 = rb0 + (hasWd ? 9 : 6);
    for (int r = 0; r < 2; ++r) {
      int ib  = (r == 0) ? rb0 : rb1;
      int rci = (r == 0) ? ci : co;
      bool wd = (r == 0) && hasWd;
      // h = relu(bn(conv1(f)))
      conv(fcur, T1, nbr3, Nl, 27, rci, co, P(ib + 0), rci);
      bnstats(T1, Nl, co, statsA);
      apply(T1, statsA, P(ib + 1), P(ib + 2), falt, Nl, co, 0,
            nullptr, nullptr, nullptr, nullptr, nullptr);
      // shortcut s = bn(f @ Wd) (identity-neighbor 1x1 conv)
      if (wd) {
        conv(fcur, T2, nullptr, Nl, 1, rci, co, P(ib + 6), rci);
        bnstats(T2, Nl, co, statsB);
      }
      // h2 = bn(conv2(h)); out = relu(h2 + s)
      conv(falt, T1, nbr3, Nl, 27, co, co, P(ib + 3), co);
      bnstats(T1, Nl, co, statsA);
      if (wd)
        apply(T1, statsA, P(ib + 4), P(ib + 5), fcur, Nl, co, 3,
              nullptr, T2, statsB, P(ib + 7), P(ib + 8));
      else
        apply(T1, statsA, P(ib + 4), P(ib + 5), fcur, Nl, co, 2,
              fcur, nullptr, nullptr, nullptr, nullptr);
    }
  }

  // ---- segment mean over batch ids at level 4 ----
  int N4 = N[4];
  zero_f32<<<dim3(2), dim3(256), 0, stream>>>(seg, 512);
  zero_f32<<<dim3(1), dim3(16), 0, stream>>>(cnt, 16);
  seg_cnt<<<dim3((N4 + 255) / 256), dim3(256), 0, stream>>>(Iv(85), cnt, N4);
  seg_acc<<<dim3((N4 * 256 + 255) / 256), dim3(256), 0, stream>>>(fcur, Iv(85), seg, N4);
  seg_fin<<<dim3(2), dim3(256), 0, stream>>>(seg, cnt, (float*)d_out, 256);
}
